// GaussianSelfAttention_7533372637348
// MI455X (gfx1250) — compile-verified, matrix operand order checked by reading, not run on hardware
//
#include <hip/hip_runtime.h>
#include <cstdint>
#include <cstddef>

// ---------------------------------------------------------------------------
// GaussianSelfAttention for MI455X (gfx1250, wave32, WMMA bf16 path)
//   B=2, G=2048, C=256, H=8, D=32
// ---------------------------------------------------------------------------

typedef unsigned short u16;
typedef __attribute__((ext_vector_type(16))) __bf16       v16bf;
typedef __attribute__((ext_vector_type(8)))  float        v8f;
typedef __attribute__((ext_vector_type(4)))  unsigned int u32x4;
typedef __attribute__((ext_vector_type(4)))  int          i32x4;

#define DEV static __device__ __forceinline__

static constexpr int BB = 2;      // batch
static constexpr int GG = 2048;   // gaussians
static constexpr int CC = 256;    // channels
static constexpr int HH = 8;      // heads
static constexpr int DD = 32;     // head dim
static constexpr int BG = BB * GG;

// -------- optional CDNA5 async global->LDS path (guarded, with fallback) ----
#if defined(__has_builtin)
#  if __has_builtin(__builtin_amdgcn_global_load_async_to_lds_b128)
#    define HAVE_ASYNC_LDS 1
#  else
#    define HAVE_ASYNC_LDS 0
#  endif
#  if __has_builtin(__builtin_amdgcn_s_wait_asynccnt)
#    define WAIT_ASYNC() __builtin_amdgcn_s_wait_asynccnt(0)
#  else
#    define WAIT_ASYNC() asm volatile("s_wait_asynccnt 0x0" ::: "memory")
#  endif
#else
#  define HAVE_ASYNC_LDS 0
#  define WAIT_ASYNC() asm volatile("s_wait_asynccnt 0x0" ::: "memory")
#endif

DEV u16 f2bf(float x) {                 // f32 -> bf16, round-to-nearest-even
  union { float f; unsigned u; } c; c.f = x;
  unsigned r = c.u + 0x7FFFu + ((c.u >> 16) & 1u);
  return (u16)(r >> 16);
}

union FragBf { v16bf v; u32x4 q[2]; u16 u[16]; };

// Load a bf16 WMMA fragment (A layout; B is fed as rows of the transposed
// operand with the same K pattern).  `row_k0` points at element (row, k0) of a
// row-major bf16 matrix where row = lane&15.  Lanes 0-15 hold
// K = {k0..k0+7, k0+16..k0+23}; lanes 16-31 hold K = {k0+8..+15, k0+24..+31}.
DEV v16bf load_frag(const u16* row_k0, int lane) {
  FragBf f;
  const u16* p = row_k0 + ((lane >> 4) << 3);
  f.q[0] = *(const u32x4*)(p);        // 16B: k .. k+7
  f.q[1] = *(const u32x4*)(p + 16);   // 16B: k+16 .. k+23
  return f.v;
}

#define WMMA_BF16(A, B, C) \
  __builtin_amdgcn_wmma_f32_16x16x32_bf16(false, (A), false, (B), (short)0, (C), false, false)

// ---------------------------------------------------------------------------
// Kernel 1: f32 -> bf16 weight conversion
// ---------------------------------------------------------------------------
__global__ void cvt_kernel(const float* __restrict__ w, u16* __restrict__ wb, int n) {
  int i = blockIdx.x * blockDim.x + threadIdx.x;
  if (i < n) wb[i] = f2bf(w[i]);
}

// ---------------------------------------------------------------------------
// Kernel 2: per-row prep: xy coords, tau = feat@tau_w.T + tau_b, feat->bf16
// One wave (32 lanes) per row; lane handles 8 consecutive channels.
// ---------------------------------------------------------------------------
__global__ __launch_bounds__(128) void prep_kernel(
    const float* __restrict__ gauss, const float* __restrict__ feat,
    const float* __restrict__ tau_w, const float* __restrict__ tau_b,
    u16* __restrict__ featb, float* __restrict__ xy, float* __restrict__ tau) {
  int lane = threadIdx.x & 31;
  int row  = blockIdx.x * 4 + (threadIdx.x >> 5);     // [0, BG)
  const float* fr = feat + (size_t)row * CC + lane * 8;

  float f[8];
#pragma unroll
  for (int j = 0; j < 8; ++j) f[j] = fr[j];

  union { u16 u[8]; u32x4 q; } pk;
#pragma unroll
  for (int j = 0; j < 8; ++j) pk.u[j] = f2bf(f[j]);
  *(u32x4*)(featb + (size_t)row * CC + lane * 8) = pk.q;

  float th[HH];
#pragma unroll
  for (int h = 0; h < HH; ++h) {
    const float* wr = tau_w + h * CC + lane * 8;
    float s = 0.f;
#pragma unroll
    for (int j = 0; j < 8; ++j) s += f[j] * wr[j];
#pragma unroll
    for (int off = 16; off; off >>= 1) s += __shfl_xor(s, off);
    th[h] = s;
  }
  if (lane == 0) {
#pragma unroll
    for (int h = 0; h < HH; ++h) tau[row * HH + h] = th[h] + tau_b[h];
    float x = gauss[(size_t)row * 11 + 0] * 102.4f - 51.2f;
    float y = gauss[(size_t)row * 11 + 1] * 102.4f - 51.2f;
    xy[row * 2 + 0] = x;
    xy[row * 2 + 1] = y;
  }
}

// ---------------------------------------------------------------------------
// Kernel 3: QKV projection.  [4096 x 256] @ [256 x 768] via WMMA bf16.
// One wave per 16x16 output tile.  Epilogue adds bias and scatters into
// Q[b,h,g,d], K[b,h,g,d], Vt[b,h,d,g] (bf16).
// ---------------------------------------------------------------------------
__global__ __launch_bounds__(128) void qkv_kernel(
    const u16* __restrict__ featb, const u16* __restrict__ wb,
    const float* __restrict__ bias,
    u16* __restrict__ qb, u16* __restrict__ kb, u16* __restrict__ vtb) {
  int lane  = threadIdx.x;                       // 0..31
  int mbase = blockIdx.x * 16;
  int nbase = (blockIdx.y * 4 + threadIdx.y) * 16;

  const u16* arow = featb + (size_t)(mbase + (lane & 15)) * CC;
  const u16* brow = wb    + (size_t)(nbase + (lane & 15)) * CC;

  v8f acc = {};
#pragma unroll
  for (int kt = 0; kt < CC / 32; ++kt) {
    v16bf a = load_frag(arow + kt * 32, lane);
    v16bf b = load_frag(brow + kt * 32, lane);
    acc = WMMA_BF16(a, b, acc);
  }

  int n  = nbase + (lane & 15);
  float bs = bias[n];
  int sec = nbase >> 8;            // 0=Q, 1=K, 2=V (uniform per wave)
  int h = (n >> 5) & 7;
  int d = n & 31;
#pragma unroll
  for (int r = 0; r < 8; ++r) {
    int gi = mbase + r + ((lane >> 4) << 3);     // global row in [0, BG)
    int b_ = gi >> 11, g = gi & (GG - 1);
    u16 bv = f2bf(acc[r] + bs);
    if (sec == 0)      qb [((size_t)(b_ * HH + h) * GG + g) * DD + d] = bv;
    else if (sec == 1) kb [((size_t)(b_ * HH + h) * GG + g) * DD + d] = bv;
    else               vtb[((size_t)(b_ * HH + h) * DD + d) * GG + g] = bv;
  }
}

// ---------------------------------------------------------------------------
// Kernel 4: fused flash attention.
// One block (8 waves) handles 128 consecutive Q rows of ONE (b,h) head, so the
// 32x32 K and Vt chunks are staged once per block into LDS (double-buffered,
// async global->LDS when available) and shared by all 8 waves: 8x less
// global/L2 traffic than per-wave fragment loads.
// ---------------------------------------------------------------------------
// Stage one 16B piece per thread of the 32x32 K chunk (threads 0-127) and the
// 32x32 Vt chunk (threads 128-255) for k rows [kbase, kbase+32).
DEV void stage_chunk(const u16* __restrict__ kh, const u16* __restrict__ vh,
                     int kbase, u16* kdst, u16* vdst, int t) {
  int q = t & 3;                 // which 16B of the 64B row
  int r = (t >> 2) & 31;        // row 0..31
  const u16* src;
  u16* dst;
  if (t < 128) { src = kh + (size_t)(kbase + r) * DD + q * 8; dst = kdst + r * 32 + q * 8; }
  else         { src = vh + (size_t)r * GG + kbase + q * 8;   dst = vdst + r * 32 + q * 8; }
#if HAVE_ASYNC_LDS
  __builtin_amdgcn_global_load_async_to_lds_b128(
      (__attribute__((address_space(1))) i32x4*)src,
      (__attribute__((address_space(3))) i32x4*)dst, 0, 0);
#else
  *(u32x4*)dst = *(const u32x4*)src;     // global_load_b128 + ds_store_b128
#endif
}

__global__ __launch_bounds__(256) void attn_kernel(
    const u16* __restrict__ qb, const u16* __restrict__ kb,
    const u16* __restrict__ vtb, const float* __restrict__ xy,
    const float* __restrict__ tau, u16* __restrict__ attnb) {
  __shared__ __align__(16) u16 kbuf[2][32 * 32];   // 2 x 2KB K chunk
  __shared__ __align__(16) u16 vbuf[2][32 * 32];   // 2 x 2KB Vt chunk
  __shared__ __align__(16) u16 pbuf[8][16 * 32];   // per-wave P tile (bf16)

  int tid  = threadIdx.x;
  int lane = tid & 31;
  int wv   = tid >> 5;
  int bh   = blockIdx.x >> 4;                      // 16 blocks per (b,h)
  int qblk = blockIdx.x & 15;
  int b    = bh >> 3;
  int h    = bh & 7;
  int qbase = qblk * 128 + wv * 16;

  const u16* kh = kb  + (size_t)bh * GG * DD;      // K [G, D]
  const u16* vh = vtb + (size_t)bh * DD * GG;      // Vt [D, G]
  v16bf qfrag = load_frag(qb + (size_t)bh * GG * DD +
                          (size_t)(qbase + (lane & 15)) * DD, lane);

  float xq[8], yq[8], tq[8], mi[8], li[8];
#pragma unroll
  for (int r = 0; r < 8; ++r) {
    int m  = qbase + r + ((lane >> 4) << 3);       // C/D layout row for vgpr r
    int gi = b * GG + m;
    xq[r] = xy[gi * 2 + 0];
    yq[r] = xy[gi * 2 + 1];
    tq[r] = tau[gi * HH + h];
    mi[r] = -__builtin_inff();
    li[r] = 0.f;
  }
  v8f O0 = {}, O1 = {};
  const float scale = 0.17677669529663687f;        // 1/sqrt(32)
  constexpr int NC = GG / 32;                      // 64 k-chunks

  stage_chunk(kh, vh, 0, kbuf[0], vbuf[0], tid);
  WAIT_ASYNC();
  __syncthreads();

  for (int kc = 0; kc < NC; ++kc) {
    int buf   = kc & 1;
    int kbase = kc * 32;
    if (kc + 1 < NC)                               // prefetch next chunk
      stage_chunk(kh, vh, kbase + 32, kbuf[buf ^ 1], vbuf[buf ^ 1], tid);

    // S = Q(16x32) * K^T(32x16) twice (k-cols 0..15 and 16..31 of chunk)
    v16bf kf0 = load_frag(&kbuf[buf][(lane & 15) * 32], lane);
    v16bf kf1 = load_frag(&kbuf[buf][(16 + (lane & 15)) * 32], lane);
    v8f z = {};
    v8f S0 = WMMA_BF16(qfrag, kf0, z);
    v8f S1 = WMMA_BF16(qfrag, kf1, z);

    int g0 = b * GG + kbase + (lane & 15);
    float xk0 = xy[g0 * 2], yk0 = xy[g0 * 2 + 1];
    float xk1 = xy[(g0 + 16) * 2], yk1 = xy[(g0 + 16) * 2 + 1];

#pragma unroll
    for (int r = 0; r < 8; ++r) {
      float dx0 = xq[r] - xk0, dy0 = yq[r] - yk0;
      float dx1 = xq[r] - xk1, dy1 = yq[r] - yk1;
      float p0 = S0[r] * scale - __fsqrt_rn(dx0 * dx0 + dy0 * dy0) * tq[r];
      float p1 = S1[r] * scale - __fsqrt_rn(dx1 * dx1 + dy1 * dy1) * tq[r];

      // row max over the 16 lanes of this half (xor<16 stays in-half)
      float mx = fmaxf(p0, p1);
#pragma unroll
      for (int off = 1; off < 16; off <<= 1) mx = fmaxf(mx, __shfl_xor(mx, off));
      float nm = fmaxf(mi[r], mx);
      float e0 = __expf(p0 - nm);
      float e1 = __expf(p1 - nm);
      float s  = e0 + e1;
#pragma unroll
      for (int off = 1; off < 16; off <<= 1) s += __shfl_xor(s, off);
      float al = __expf(mi[r] - nm);
      li[r] = li[r] * al + s;
      mi[r] = nm;
      O0[r] *= al;
      O1[r] *= al;

      int lm = r + ((lane >> 4) << 3);             // local P row
      pbuf[wv][lm * 32 + (lane & 15)]      = f2bf(e0);
      pbuf[wv][lm * 32 + (lane & 15) + 16] = f2bf(e1);
    }

    // O(16x32) += P(16x32) * V(32x32):  P re-read from LDS in A layout,
    // V chunk read from the shared LDS stage.
    v16bf pf  = load_frag(&pbuf[wv][(lane & 15) * 32], lane);
    v16bf vf0 = load_frag(&vbuf[buf][(lane & 15) * 32], lane);
    v16bf vf1 = load_frag(&vbuf[buf][(16 + (lane & 15)) * 32], lane);
    O0 = WMMA_BF16(pf, vf0, O0);
    O1 = WMMA_BF16(pf, vf1, O1);

    WAIT_ASYNC();          // next chunk fully in LDS (async path)
    __syncthreads();       // all waves done with `buf` before it is re-staged
  }

#pragma unroll
  for (int r = 0; r < 8; ++r) {
    int m = qbase + r + ((lane >> 4) << 3);
    float inv = 1.f / li[r];
    size_t o = ((size_t)b * GG + m) * CC + h * DD;
    attnb[o + (lane & 15)]      = f2bf(O0[r] * inv);
    attnb[o + 16 + (lane & 15)] = f2bf(O1[r] * inv);
  }
}

// ---------------------------------------------------------------------------
// Kernel 5: output projection.  [4096 x 256] @ [256 x 256] + bias -> f32 out
// ---------------------------------------------------------------------------
__global__ __launch_bounds__(128) void outproj_kernel(
    const u16* __restrict__ attnb, const u16* __restrict__ wb,
    const float* __restrict__ bias, float* __restrict__ out) {
  int lane  = threadIdx.x;
  int mbase = blockIdx.x * 16;
  int nbase = (blockIdx.y * 4 + threadIdx.y) * 16;

  const u16* arow = attnb + (size_t)(mbase + (lane & 15)) * CC;
  const u16* brow = wb    + (size_t)(nbase + (lane & 15)) * CC;

  v8f acc = {};
#pragma unroll
  for (int kt = 0; kt < CC / 32; ++kt) {
    v16bf a = load_frag(arow + kt * 32, lane);
    v16bf b = load_frag(brow + kt * 32, lane);
    acc = WMMA_BF16(a, b, acc);
  }
  int n = nbase + (lane & 15);
  float bs = bias[n];
#pragma unroll
  for (int r = 0; r < 8; ++r) {
    int gi = mbase + r + ((lane >> 4) << 3);
    out[(size_t)gi * CC + n] = acc[r] + bs;
  }
}

// ---------------------------------------------------------------------------
// Launch
// ---------------------------------------------------------------------------
extern "C" void kernel_launch(void* const* d_in, const int* in_sizes, int n_in,
                              void* d_out, int out_size, void* d_ws, size_t ws_size,
                              hipStream_t stream) {
  const float* gauss      = (const float*)d_in[0];   // [B,G,11]
  const float* feat       = (const float*)d_in[1];   // [B,G,C]
  const float* tau_w      = (const float*)d_in[2];   // [H,C]
  const float* tau_b      = (const float*)d_in[3];   // [H]
  const float* in_proj_w  = (const float*)d_in[4];   // [3C,C]
  const float* in_proj_b  = (const float*)d_in[5];   // [3C]
  const float* out_proj_w = (const float*)d_in[6];   // [C,C]
  const float* out_proj_b = (const float*)d_in[7];   // [C]
  float* out = (float*)d_out;

  char* ws = (char*)d_ws;
  u16*   featb = (u16*)(ws);                         // 2 MB
  u16*   qb    = (u16*)(ws + (2u << 20));            // 2 MB
  u16*   kb    = (u16*)(ws + (4u << 20));            // 2 MB
  u16*   vtb   = (u16*)(ws + (6u << 20));            // 2 MB
  u16*   attnb = (u16*)(ws + (8u << 20));            // 2 MB
  u16*   wqkvb = (u16*)(ws + (10u << 20));           // 384 KB
  u16*   woutb = (u16*)(ws + (10u << 20) + 393216);  // 128 KB
  float* xy    = (float*)(ws + (10u << 20) + 524288);          // 32 KB
  float* tau   = (float*)(ws + (10u << 20) + 524288 + 32768);  // 128 KB

  cvt_kernel<<<(3 * CC * CC + 255) / 256, 256, 0, stream>>>(in_proj_w, wqkvb, 3 * CC * CC);
  cvt_kernel<<<(CC * CC + 255) / 256, 256, 0, stream>>>(out_proj_w, woutb, CC * CC);
  prep_kernel<<<BG / 4, 128, 0, stream>>>(gauss, feat, tau_w, tau_b, featb, xy, tau);
  qkv_kernel<<<dim3(BG / 16, (3 * CC / 16) / 4), dim3(32, 4), 0, stream>>>(
      featb, wqkvb, in_proj_b, qb, kb, vtb);
  attn_kernel<<<BB * HH * 16, 256, 0, stream>>>(qb, kb, vtb, xy, tau, attnb);
  outproj_kernel<<<dim3(BG / 16, (CC / 16) / 4), dim3(32, 4), 0, stream>>>(
      attnb, woutb, out_proj_b, out);
}
